// MultiHeadAttention_52252572123552
// MI455X (gfx1250) — matrix operand order repaired and checked by reading, not compiled, not run on hardware
//
#include <hip/hip_runtime.h>

// ---------------------------------------------------------------------------
// MI455X (gfx1250) multi-head attention forward, bf16 WMMA everywhere.
//   D=1024, H=16, dk=64, B=2, S=2048.
// GEMMs: v_wmma_f32_16x16x32_bf16. Contiguous tile staging uses
// global_load_async_to_lds_b128 (ASYNCcnt); transposed stages use ds stores.
// Attention scale 1/sqrt(64)=1/8 folded into Q (exact exponent shift in bf16).
// ---------------------------------------------------------------------------

typedef __bf16 bf16;
typedef __attribute__((ext_vector_type(16))) __bf16 v16bf;
typedef __attribute__((ext_vector_type(8)))  __bf16 v8bf;
typedef __attribute__((ext_vector_type(8)))  float  v8f;

#define D_MODEL 1024
#define NHEAD   16
#define DK      64
#define SEQ     2048
#define BATCH   2
#define NTOK    (BATCH * SEQ)   // 4096

__device__ __forceinline__ bf16 f2bf(float f) {
  union { float f; unsigned u; } v; v.f = f;
  unsigned r = (v.u + 0x7FFFu + ((v.u >> 16) & 1u)) >> 16;   // round-nearest-even
  union { unsigned short s; bf16 b; } o; o.s = (unsigned short)r;
  return o.b;
}

__device__ __forceinline__ v8f wmma_bf16(v16bf a, v16bf b, v8f c) {
  // (neg_a, A, neg_b, B, c_mod, C, reuse_a, reuse_b)
  return __builtin_amdgcn_wmma_f32_16x16x32_bf16(false, a, false, b, (short)0, c,
                                                 false, false);
}

// A-matrix fragment gather (16x32 bf16). `row` = start of this lane's M row in
// a row-major [M][K>=32] tile. Per ISA layout: lanes 0-15 hold K={0..7,16..23},
// lanes 16-31 hold K={8..15,24..31}.
__device__ __forceinline__ v16bf a_frag(const bf16* row, int lane) {
  const int k0 = (lane & 16) ? 8 : 0;
  v8bf lo = *(const v8bf*)(row + k0);
  v8bf hi = *(const v8bf*)(row + k0 + 16);
  return __builtin_shufflevector(lo, hi, 0, 1, 2, 3, 4, 5, 6, 7,
                                          8, 9, 10, 11, 12, 13, 14, 15);
}
// B-matrix fragment (32x16 bf16): lane&15 = N column, K order matches a_frag;
// `col` = &tile[0][n] of a [K][N] tile with row stride `ld`.
__device__ __forceinline__ v16bf b_frag(const bf16* col, int ld, int lane) {
  const int k0 = (lane & 16) ? 8 : 0;
  v16bf b;
#pragma unroll
  for (int s = 0; s < 8; ++s) {
    b[s] = col[(k0 + s) * ld];
    b[8 + s] = col[(k0 + 16 + s) * ld];
  }
  return b;
}

// --- gfx1250 async global->LDS copy (16 bytes per lane, ASYNCcnt-tracked) ---
// Generic LDS pointer -> LDS byte address = low 32 bits (ISA flat aperture).
__device__ __forceinline__ void async_ld16(void* lds_ptr, const void* gptr) {
  unsigned l = (unsigned)(unsigned long long)(uintptr_t)lds_ptr;
  asm volatile("global_load_async_to_lds_b128 %0, %1, off"
               :: "v"(l), "v"(gptr) : "memory");
}
__device__ __forceinline__ void wait_async0() {
  asm volatile("s_wait_asynccnt 0x0" ::: "memory");
}

// ---------------------------------------------------------------------------
// fp32 -> bf16 cast
// ---------------------------------------------------------------------------
__global__ void cvt_bf16_kernel(const float* __restrict__ in,
                                bf16* __restrict__ out, int n) {
  int i = blockIdx.x * 256 + threadIdx.x;
  if (i < n) out[i] = f2bf(in[i]);
}

// ---------------------------------------------------------------------------
// QKV projection + bias + RoPE.  grid = (N/64, M/64, 3), block = 128 (4 waves)
// Out = X * W^T + b (NT gemm); RoPE on Q/K; store bf16 as [B, H, S, dk].
// ---------------------------------------------------------------------------
__global__ __launch_bounds__(128) void qkv_rope_kernel(
    const bf16* __restrict__ X,
    const bf16* __restrict__ Wq, const bf16* __restrict__ Wk,
    const bf16* __restrict__ Wv,
    const float* __restrict__ bq, const float* __restrict__ bk,
    const float* __restrict__ bv,
    bf16* __restrict__ Q, bf16* __restrict__ K, bf16* __restrict__ V) {
  const int which = blockIdx.z;  // 0=Q 1=K 2=V
  const int mtile = blockIdx.y * 64;
  const int ntile = blockIdx.x * 64;
  const int t = threadIdx.x, wave = t >> 5, lane = t & 31;

  const bf16*  W    = (which == 0) ? Wq : (which == 1) ? Wk : Wv;
  const float* bias = (which == 0) ? bq : (which == 1) ? bk : bv;
  bf16*        Out  = (which == 0) ? Q  : (which == 1) ? K  : V;

  __shared__ __align__(16) bf16 sX[64][32];    // [m][k] row-major (A tiles)
  __shared__ __align__(16) bf16 sWt[32][64];   // [k][n] transposed (B tiles)

  v8f acc[4] = {};

  for (int k0 = 0; k0 < D_MODEL; k0 += 32) {
    {  // X tile: 64 rows x 32 k — contiguous, async copy to LDS
      int r = t >> 1, c = (t & 1) * 16;
      const bf16* src = X + (size_t)(mtile + r) * D_MODEL + k0 + c;
      async_ld16(&sX[r][c], src);
      async_ld16(&sX[r][c + 8], src + 8);
    }
    // W tile transposed into [k][n] (element scatter; ds stores)
    for (int ch = t; ch < 256; ch += 128) {
      int n = ch >> 2, kc = (ch & 3) * 8;
      v8bf w = *(const v8bf*)(W + (size_t)(ntile + n) * D_MODEL + k0 + kc);
#pragma unroll
      for (int j = 0; j < 8; ++j) sWt[kc + j][n] = w[j];
    }
    // prefetch next k-step tiles into cache while this one computes
    if (k0 + 32 < D_MODEL) {
      __builtin_prefetch(X + (size_t)(mtile + (t >> 1)) * D_MODEL + k0 + 32, 0, 3);
      __builtin_prefetch(W + (size_t)(ntile + (t >> 1)) * D_MODEL + k0 + 32, 0, 3);
    }
    wait_async0();
    __syncthreads();

    v16bf a = a_frag(&sX[wave * 16 + (lane & 15)][0], lane);
#pragma unroll
    for (int nf = 0; nf < 4; ++nf) {
      v16bf b = b_frag(&sWt[0][nf * 16 + (lane & 15)], 64, lane);  // lane&15 = N
      acc[nf] = wmma_bf16(a, b, acc[nf]);
    }
    __syncthreads();
  }

  // Epilogue: bias, RoPE, scatter to [B,H,S,dk].
  // Accumulator: lane holds column n=(lane&15) of fragment nf, rows
  // m = (lane>>4)*8 + r in vgpr r.
#pragma unroll
  for (int nf = 0; nf < 4; ++nf) {
    const int n  = ntile + nf * 16 + (lane & 15);
    const float bn = bias[n];
#pragma unroll
    for (int r = 0; r < 8; ++r) {
      const int g  = mtile + wave * 16 + (lane >> 4) * 8 + r;  // token row
      const int b_ = g >> 11, s_ = g & (SEQ - 1);
      const int h  = n >> 6, d_ = n & (DK - 1);
      float val = acc[nf][r] + bn;
      if (which < 2) {
        // RoPE pair partner lives in lane^1 (n differs only in bit 0)
        float pv = __shfl_xor(val, 1, 32);
        const int p = d_ >> 1;
        float ang = (float)s_ * __expf(-(float)p * 0.28782313662f);  // 1e4^(-p/32)
        float sn, cs;
        sincosf(ang, &sn, &cs);
        val = (d_ & 1) ? (val * cs + pv * sn) : (val * cs - pv * sn);
        if (which == 0) val *= 0.125f;  // fold 1/sqrt(dk) into Q (exact in bf16)
      }
      Out[(((size_t)b_ * NHEAD + h) * SEQ + s_) * DK + d_] = f2bf(val);
    }
  }
}

// ---------------------------------------------------------------------------
// Flash attention.  grid = (S/64, B*H), block = 128 (4 waves).
// Each wave owns 16 q rows; KV streamed in 64-wide tiles through LDS.
// Online softmax in f32; P re-gathered as bf16 A fragments via LDS.
// Output written bf16 token-major as [B, S, H*dk].
// ---------------------------------------------------------------------------
__global__ __launch_bounds__(128) void attention_kernel(
    const bf16* __restrict__ Q, const bf16* __restrict__ K,
    const bf16* __restrict__ V, bf16* __restrict__ O) {
  const int bh    = blockIdx.y;
  const int qtile = blockIdx.x * 64;
  const int t = threadIdx.x, wave = t >> 5, lane = t & 31;
  const size_t base = (size_t)bh * SEQ * DK;

  __shared__ __align__(16) bf16 sKt[64][64];     // [d][tok]  (B tile for QK^T)
  __shared__ __align__(16) bf16 sV[64][64];      // [tok][d]  (B tile for P.V)
  __shared__ __align__(16) bf16 sP[4][16][64];   // wave-private P round-trip

  // Q fragments for this wave's 16 rows (Q already scaled by 1/8)
  const bf16* qrow =
      Q + base + (size_t)(qtile + wave * 16 + (lane & 15)) * DK;
  v16bf aq0 = a_frag(qrow, lane);
  v16bf aq1 = a_frag(qrow + 32, lane);

  float m_i[8], l_i[8];
#pragma unroll
  for (int r = 0; r < 8; ++r) { m_i[r] = -3.0e38f; l_i[r] = 0.f; }
  v8f o_acc[4] = {};

  for (int kt = 0; kt < SEQ; kt += 64) {
    // V tile: contiguous 8 KB — async copy straight to LDS
    {
      const bf16* gv = V + base + (size_t)kt * DK;
      bf16* sv = &sV[0][0];
      for (int i = t; i < 512; i += 128) async_ld16(sv + i * 8, gv + i * 8);
    }
    // K tile transposed into [d][tok] (element scatter; ds stores)
    for (int ch = t; ch < 512; ch += 128) {
      int tok = ch >> 3, kc = (ch & 7) * 8;
      v8bf kk = *(const v8bf*)(K + base + (size_t)(kt + tok) * DK + kc);
#pragma unroll
      for (int j = 0; j < 8; ++j) sKt[kc + j][tok] = kk[j];
    }
    // prefetch next KV tile rows
    if (kt + 64 < SEQ) {
      __builtin_prefetch(K + base + (size_t)(kt + 64 + (t >> 1)) * DK, 0, 3);
      __builtin_prefetch(V + base + (size_t)(kt + 64 + (t >> 1)) * DK, 0, 3);
    }
    wait_async0();
    __syncthreads();

    // S = Q K^T : 4 fragments of 16 kv columns, K-dim = dk = 2 x 32
    v8f sc[4];
#pragma unroll
    for (int nf = 0; nf < 4; ++nf) {
      v8f s = {};
      s = wmma_bf16(aq0, b_frag(&sKt[0][nf * 16 + (lane & 15)], 64, lane), s);
      s = wmma_bf16(aq1, b_frag(&sKt[32][nf * 16 + (lane & 15)], 64, lane), s);
      sc[nf] = s;
    }

    // Online softmax; row m spans 16 lanes of one half-wave -> butterfly 1..8
#pragma unroll
    for (int r = 0; r < 8; ++r) {
      float mx = fmaxf(fmaxf(sc[0][r], sc[1][r]), fmaxf(sc[2][r], sc[3][r]));
#pragma unroll
      for (int off = 8; off > 0; off >>= 1) mx = fmaxf(mx, __shfl_xor(mx, off, 32));
      float m_new = fmaxf(m_i[r], mx);
      float scale = __expf(m_i[r] - m_new);
      float rs = 0.f;
#pragma unroll
      for (int nf = 0; nf < 4; ++nf) {
        float p = __expf(sc[nf][r] - m_new);
        rs += p;
        sP[wave][(lane >> 4) * 8 + r][nf * 16 + (lane & 15)] = f2bf(p);
      }
#pragma unroll
      for (int off = 8; off > 0; off >>= 1) rs += __shfl_xor(rs, off, 32);
      l_i[r] = l_i[r] * scale + rs;
      m_i[r] = m_new;
#pragma unroll
      for (int df = 0; df < 4; ++df) o_acc[df][r] *= scale;
    }

    // Re-gather P as A fragments (same-wave LDS ops are in-order)
    const bf16* prow = &sP[wave][lane & 15][0];
    v16bf ap0 = a_frag(prow, lane);
    v16bf ap1 = a_frag(prow + 32, lane);

    // O += P * V ; V tile is already [tok][d] => lane&15 = d = N index
#pragma unroll
    for (int df = 0; df < 4; ++df) {
      o_acc[df] = wmma_bf16(ap0, b_frag(&sV[0][df * 16 + (lane & 15)], 64, lane), o_acc[df]);
      o_acc[df] = wmma_bf16(ap1, b_frag(&sV[32][df * 16 + (lane & 15)], 64, lane), o_acc[df]);
    }
    __syncthreads();
  }

  // Normalize and store token-major [B, S, H*dk]
  const int b_ = bh >> 4, h = bh & 15;
#pragma unroll
  for (int r = 0; r < 8; ++r) {
    const int s_ = qtile + wave * 16 + (lane >> 4) * 8 + r;
    float inv = 1.f / l_i[r];
#pragma unroll
    for (int df = 0; df < 4; ++df) {
      O[((size_t)b_ * SEQ + s_) * D_MODEL + h * DK + df * 16 + (lane & 15)] =
          f2bf(o_acc[df][r] * inv);
    }
  }
}

// ---------------------------------------------------------------------------
// Output projection: Y = A * Wo^T + bo, fp32 result. grid=(N/64, M/64), blk=128
// ---------------------------------------------------------------------------
__global__ __launch_bounds__(128) void out_proj_kernel(
    const bf16* __restrict__ A, const bf16* __restrict__ W,
    const float* __restrict__ bias, float* __restrict__ Y) {
  const int mtile = blockIdx.y * 64, ntile = blockIdx.x * 64;
  const int t = threadIdx.x, wave = t >> 5, lane = t & 31;
  __shared__ __align__(16) bf16 sX[64][32];
  __shared__ __align__(16) bf16 sWt[32][64];
  v8f acc[4] = {};

  for (int k0 = 0; k0 < D_MODEL; k0 += 32) {
    {
      int r = t >> 1, c = (t & 1) * 16;
      const bf16* src = A + (size_t)(mtile + r) * D_MODEL + k0 + c;
      async_ld16(&sX[r][c], src);
      async_ld16(&sX[r][c + 8], src + 8);
    }
    for (int ch = t; ch < 256; ch += 128) {
      int n = ch >> 2, kc = (ch & 3) * 8;
      v8bf w = *(const v8bf*)(W + (size_t)(ntile + n) * D_MODEL + k0 + kc);
#pragma unroll
      for (int j = 0; j < 8; ++j) sWt[kc + j][n] = w[j];
    }
    if (k0 + 32 < D_MODEL) {
      __builtin_prefetch(A + (size_t)(mtile + (t >> 1)) * D_MODEL + k0 + 32, 0, 3);
      __builtin_prefetch(W + (size_t)(ntile + (t >> 1)) * D_MODEL + k0 + 32, 0, 3);
    }
    wait_async0();
    __syncthreads();

    v16bf a = a_frag(&sX[wave * 16 + (lane & 15)][0], lane);
#pragma unroll
    for (int nf = 0; nf < 4; ++nf)
      acc[nf] = wmma_bf16(a, b_frag(&sWt[0][nf * 16 + (lane & 15)], 64, lane), acc[nf]);
    __syncthreads();
  }

#pragma unroll
  for (int nf = 0; nf < 4; ++nf) {
    int n = ntile + nf * 16 + (lane & 15);
    float bn = bias[n];
#pragma unroll
    for (int r = 0; r < 8; ++r) {
      int g = mtile + wave * 16 + (lane >> 4) * 8 + r;
      Y[(size_t)g * D_MODEL + n] = acc[nf][r] + bn;
    }
  }
}

// ---------------------------------------------------------------------------
extern "C" void kernel_launch(void* const* d_in, const int* in_sizes, int n_in,
                              void* d_out, int out_size, void* d_ws,
                              size_t ws_size, hipStream_t stream) {
  const float* x  = (const float*)d_in[0];
  const float* wq = (const float*)d_in[1];
  const float* bq = (const float*)d_in[2];
  const float* wk = (const float*)d_in[3];
  const float* bk = (const float*)d_in[4];
  const float* wv = (const float*)d_in[5];
  const float* bv = (const float*)d_in[6];
  const float* wo = (const float*)d_in[7];
  const float* bo = (const float*)d_in[8];
  float* out = (float*)d_out;

  // Workspace layout (bf16):  ~48 MB total
  bf16* Xb  = (bf16*)d_ws;
  bf16* Wqb = Xb  + (size_t)NTOK * D_MODEL;
  bf16* Wkb = Wqb + (size_t)D_MODEL * D_MODEL;
  bf16* Wvb = Wkb + (size_t)D_MODEL * D_MODEL;
  bf16* Wob = Wvb + (size_t)D_MODEL * D_MODEL;
  bf16* Qb  = Wob + (size_t)D_MODEL * D_MODEL;
  bf16* Kb  = Qb  + (size_t)NTOK * D_MODEL;
  bf16* Vb  = Kb  + (size_t)NTOK * D_MODEL;
  bf16* Ob  = Vb  + (size_t)NTOK * D_MODEL;

  const int nx = NTOK * D_MODEL;
  const int nw = D_MODEL * D_MODEL;
  cvt_bf16_kernel<<<(nx + 255) / 256, 256, 0, stream>>>(x, Xb, nx);
  cvt_bf16_kernel<<<(nw + 255) / 256, 256, 0, stream>>>(wq, Wqb, nw);
  cvt_bf16_kernel<<<(nw + 255) / 256, 256, 0, stream>>>(wk, Wkb, nw);
  cvt_bf16_kernel<<<(nw + 255) / 256, 256, 0, stream>>>(wv, Wvb, nw);
  cvt_bf16_kernel<<<(nw + 255) / 256, 256, 0, stream>>>(wo, Wob, nw);

  qkv_rope_kernel<<<dim3(D_MODEL / 64, NTOK / 64, 3), 128, 0, stream>>>(
      Xb, Wqb, Wkb, Wvb, bq, bk, bv, Qb, Kb, Vb);

  attention_kernel<<<dim3(SEQ / 64, BATCH * NHEAD), 128, 0, stream>>>(Qb, Kb,
                                                                      Vb, Ob);

  out_proj_kernel<<<dim3(D_MODEL / 64, NTOK / 64), 128, 0, stream>>>(Ob, Wob,
                                                                     bo, out);
}
